// HilbertDilatedAttentionTriton_58926951301485
// MI455X (gfx1250) — matrix operand order repaired and checked by reading, not compiled, not run on hardware
//
#include <hip/hip_runtime.h>
#include <stdint.h>

// ---------------------------------------------------------------------------
// CDNA5 (gfx1250) fused Hilbert-dilated attention, bf16 WMMA pipeline with
// async global->LDS double-buffered staging.
// ---------------------------------------------------------------------------

typedef __attribute__((ext_vector_type(16))) __bf16 v16bf;
typedef __attribute__((ext_vector_type(8)))  float  v8f;

#define WMMA_BF16(a, b, c) \
  __builtin_amdgcn_wmma_f32_16x16x32_bf16(false, (a), false, (b), (short)0, (c), false, false)

// Async copy of 16 bytes global -> LDS (per-lane addresses), ASYNCcnt-tracked.
#define ASYNC_CP16(ldsaddr, gaddr)                                   \
  asm volatile("global_load_async_to_lds_b128 %0, %1, off"           \
               :: "v"(ldsaddr), "v"(gaddr) : "memory")
#define WAIT_ASYNC0() asm volatile("s_wait_asynccnt 0x0" ::: "memory")

__device__ __forceinline__ unsigned short f2bf(float f) {
  unsigned int u = __float_as_uint(f);
  u += 0x7FFFu + ((u >> 16) & 1u);   // round to nearest even
  return (unsigned short)(u >> 16);
}

__device__ __forceinline__ v8f vzero() {
  v8f z = {0.f, 0.f, 0.f, 0.f, 0.f, 0.f, 0.f, 0.f};
  return z;
}

union Frag {
  v16bf v;
  unsigned int u32[8];
  unsigned short u16[16];
};

// ---------------------------------------------------------------------------
// fp32 -> bf16 conversion
// ---------------------------------------------------------------------------
__global__ void cvt_f32_bf16(const float* __restrict__ in,
                             unsigned short* __restrict__ out, int n) {
  int i = blockIdx.x * blockDim.x + threadIdx.x;
  if (i < n) out[i] = f2bf(in[i]);
}

// hk[j] = hilbert_map[4*j]   (pos % 256 % 4 == 0  <=>  pos % 4 == 0)
__global__ void build_hk(const int* __restrict__ hm, int* __restrict__ hk) {
  hk[threadIdx.x] = hm[4 * threadIdx.x];
}

// ---------------------------------------------------------------------------
// Tiled bf16 WMMA GEMM core: C[m][n] = sum_k A[m][k] * B[n][k]
// BM=BN=128, BK=32, 256 threads (8 waves), wave grid 2(M) x 4(N),
// each wave: 4 m-tiles x 2 n-tiles of 16x16. Double-buffered async staging.
// ---------------------------------------------------------------------------
#define BM 128
#define BN 128
#define BK 32
#define NKT 32   // K / BK with K = 1024
#define LDT 40   // padded LDS stride (bf16 units); 80B rows, 16B aligned

// Stage one 128x32 A-row-block and 128x32 B-row-block into LDS buffer `buf`
// via async copies: thread t<128 copies A row t, t>=128 copies B row t-128.
__device__ __forceinline__ void gemm_stage_async(
    const unsigned short* __restrict__ A, const unsigned short* __restrict__ Bm,
    unsigned short (*As)[BM][LDT], unsigned short (*Bs)[BM][LDT],
    int bm, int bn, int kt, int buf, int t) {
  const int row = t & 127;
  const unsigned short* gsrc = (t < 128)
      ? (A  + (size_t)(bm + row) * 1024 + kt)
      : (Bm + (size_t)(bn + row) * 1024 + kt);
  unsigned short* ldst = (t < 128) ? &As[buf][row][0] : &Bs[buf][row][0];
  const unsigned lb = (unsigned)(uintptr_t)ldst;
  const unsigned long long g = (unsigned long long)(uintptr_t)gsrc;
#pragma unroll
  for (int i = 0; i < 4; i++) ASYNC_CP16(lb + 16u * i, g + 16ull * i);
}

// QKV projection: A = x_bf16 [8192,1024], B = w_qkv_bf16 [3072,1024]
// epilogue scatters into q/k/v [b(2)][h(16)][s(4096)][d(64)] bf16.
__global__ __launch_bounds__(256)
void gemm_qkv(const unsigned short* __restrict__ A,
              const unsigned short* __restrict__ Bm,
              unsigned short* __restrict__ qb,
              unsigned short* __restrict__ kb,
              unsigned short* __restrict__ vb) {
  __shared__ unsigned short As[2][BM][LDT];
  __shared__ unsigned short Bs[2][BM][LDT];
  const int t = threadIdx.x;
  const int lane = t & 31, wid = t >> 5;
  const int waveM = wid >> 2, waveN = wid & 3;
  const int bm = blockIdx.x * BM, bn = blockIdx.y * BN;
  const int r = lane & 15, half = lane >> 4;

  v8f acc[4][2];
#pragma unroll
  for (int i = 0; i < 4; i++)
#pragma unroll
    for (int j = 0; j < 2; j++) acc[i][j] = vzero();

  gemm_stage_async(A, Bm, As, Bs, bm, bn, 0, 0, t);

  for (int it = 0; it < NKT; ++it) {
    WAIT_ASYNC0();       // tile `it` landed in LDS (this wave's copies)
    __syncthreads();     // all waves' copies visible; prev buffer reads done
    const int cur = it & 1;
    if (it + 1 < NKT)    // overlap next tile's copy with this tile's WMMAs
      gemm_stage_async(A, Bm, As, Bs, bm, bn, (it + 1) * BK, cur ^ 1, t);

    Frag af[4], bf[2];
#pragma unroll
    for (int mt = 0; mt < 4; mt++) {
      const int row = waveM * 64 + mt * 16 + r;
#pragma unroll
      for (int i = 0; i < 8; i++) {
        const int k0 = (i < 4) ? (half * 8 + 2 * i) : (16 + half * 8 + 2 * (i - 4));
        af[mt].u32[i] = *(const unsigned int*)&As[cur][row][k0];
      }
    }
#pragma unroll
    for (int nt = 0; nt < 2; nt++) {
      const int row = waveN * 32 + nt * 16 + r;
#pragma unroll
      for (int i = 0; i < 8; i++) {
        bf[nt].u32[i] = *(const unsigned int*)&Bs[cur][row][(i < 4) ? (half * 8 + 2 * i) : (16 + half * 8 + 2 * (i - 4))];
      }
    }
#pragma unroll
    for (int mt = 0; mt < 4; mt++)
#pragma unroll
      for (int nt = 0; nt < 2; nt++)
        acc[mt][nt] = WMMA_BF16(af[mt].v, bf[nt].v, acc[mt][nt]);
  }

  // Epilogue: route columns into q/k/v per-head layout.
#pragma unroll
  for (int mt = 0; mt < 4; mt++) {
#pragma unroll
    for (int nt = 0; nt < 2; nt++) {
      const int nb = bn + waveN * 32 + nt * 16;   // 16-aligned: c,h uniform in tile
      const int c = nb >> 10;
      const int h = (nb >> 6) & 15;
      const int dbase = nb & 63;
      unsigned short* dstbuf = (c == 0) ? qb : ((c == 1) ? kb : vb);
#pragma unroll
      for (int j = 0; j < 8; j++) {
        const int m = bm + waveM * 64 + mt * 16 + half * 8 + j;
        const int b = m >> 12, s = m & 4095;
        const size_t idx = (((size_t)(b * 16 + h)) * 4096 + s) * 64 + dbase + r;
        dstbuf[idx] = f2bf(acc[mt][nt][j]);
      }
    }
  }
}

// Output projection: A = ctx_perm bf16 [8192,1024], B = w_out_bf16 [1024,1024]
// epilogue writes fp32 d_out [8192,1024].
__global__ __launch_bounds__(256)
void gemm_out(const unsigned short* __restrict__ A,
              const unsigned short* __restrict__ Bm,
              float* __restrict__ out) {
  __shared__ unsigned short As[2][BM][LDT];
  __shared__ unsigned short Bs[2][BM][LDT];
  const int t = threadIdx.x;
  const int lane = t & 31, wid = t >> 5;
  const int waveM = wid >> 2, waveN = wid & 3;
  const int bm = blockIdx.x * BM, bn = blockIdx.y * BN;
  const int r = lane & 15, half = lane >> 4;

  v8f acc[4][2];
#pragma unroll
  for (int i = 0; i < 4; i++)
#pragma unroll
    for (int j = 0; j < 2; j++) acc[i][j] = vzero();

  gemm_stage_async(A, Bm, As, Bs, bm, bn, 0, 0, t);

  for (int it = 0; it < NKT; ++it) {
    WAIT_ASYNC0();
    __syncthreads();
    const int cur = it & 1;
    if (it + 1 < NKT)
      gemm_stage_async(A, Bm, As, Bs, bm, bn, (it + 1) * BK, cur ^ 1, t);

    Frag af[4], bf[2];
#pragma unroll
    for (int mt = 0; mt < 4; mt++) {
      const int row = waveM * 64 + mt * 16 + r;
#pragma unroll
      for (int i = 0; i < 8; i++) {
        const int k0 = (i < 4) ? (half * 8 + 2 * i) : (16 + half * 8 + 2 * (i - 4));
        af[mt].u32[i] = *(const unsigned int*)&As[cur][row][k0];
      }
    }
#pragma unroll
    for (int nt = 0; nt < 2; nt++) {
      const int row = waveN * 32 + nt * 16 + r;
#pragma unroll
      for (int i = 0; i < 8; i++) {
        bf[nt].u32[i] = *(const unsigned int*)&Bs[cur][row][(i < 4) ? (half * 8 + 2 * i) : (16 + half * 8 + 2 * (i - 4))];
      }
    }
#pragma unroll
    for (int mt = 0; mt < 4; mt++)
#pragma unroll
      for (int nt = 0; nt < 2; nt++)
        acc[mt][nt] = WMMA_BF16(af[mt].v, bf[nt].v, acc[mt][nt]);
  }

#pragma unroll
  for (int mt = 0; mt < 4; mt++) {
#pragma unroll
    for (int nt = 0; nt < 2; nt++) {
      const int n = bn + waveN * 32 + nt * 16 + r;
#pragma unroll
      for (int j = 0; j < 8; j++) {
        const int m = bm + waveM * 64 + mt * 16 + half * 8 + j;
        out[(size_t)m * 1024 + n] = acc[mt][nt][j];
      }
    }
  }
}

// ---------------------------------------------------------------------------
// Fused flash attention over 1024 dilated+Hilbert-gathered keys.
// Block = 128 threads (4 waves); wave owns 16 query rows; 32-key chunks.
// K tile staged via async global->LDS; V tile transposed through VGPRs.
// ---------------------------------------------------------------------------
__global__ __launch_bounds__(128)
void attn_fused(const unsigned short* __restrict__ qb,
                const unsigned short* __restrict__ kb,
                const unsigned short* __restrict__ vb,
                const int* __restrict__ hk,
                const int* __restrict__ hm,
                unsigned short* __restrict__ ctx_perm) {
  __shared__ unsigned short ks[32][64];     // 32 keys x 64 dims, row-major
  __shared__ unsigned short vT[64][36];     // transposed V: [d][key], padded
  __shared__ unsigned short pl[4][16][36];  // per-wave P tile (16 rows x 32 keys)

  const int t = threadIdx.x;
  const int lane = t & 31, w = t >> 5;
  const int r = lane & 15, half = lane >> 4;
  const int bh = blockIdx.y;            // b*16 + h
  const int b = bh >> 4, h = bh & 15;
  const int qbase = blockIdx.x * 64 + w * 16;

  // Q fragments for this wave's 16 rows (d = 0..31, 32..63), held in registers.
  const unsigned short* qrow = qb + ((size_t)bh * 4096 + qbase + r) * 64;
  Frag qa[2];
#pragma unroll
  for (int dstep = 0; dstep < 2; dstep++)
#pragma unroll
    for (int i = 0; i < 8; i++) {
      const int k0 = dstep * 32 +
          ((i < 4) ? (half * 8 + 2 * i) : (16 + half * 8 + 2 * (i - 4)));
      qa[dstep].u32[i] = *(const unsigned int*)(qrow + k0);
    }

  v8f ctx[4];
#pragma unroll
  for (int nt = 0; nt < 4; nt++) ctx[nt] = vzero();
  float mr[8], Ls[8];
#pragma unroll
  for (int j = 0; j < 8; j++) { mr[j] = -3.0e38f; Ls[j] = 0.f; }

  const size_t kvbase = (size_t)bh * 4096 * 64;

  for (int ch = 0; ch < 32; ch++) {
    // Stage 32 gathered K rows (async, row-major) and V rows (transposed).
    {
      const int row = t >> 2, seg = t & 3;
      const int gj = hk[ch * 32 + row];
      // K: 32B per thread, straight copy -> async global->LDS path.
      {
        const unsigned lb =
            (unsigned)(uintptr_t)&ks[row][seg * 16];
        const unsigned long long g =
            (unsigned long long)(uintptr_t)(kb + kvbase + (size_t)gj * 64 + seg * 16);
        ASYNC_CP16(lb, g);
        ASYNC_CP16(lb + 16u, g + 16ull);
      }
      // V: needs transpose, stays VGPR-mediated.
      Frag vv;
      const uint4* vsrc = (const uint4*)(vb + kvbase + (size_t)gj * 64 + seg * 16);
      ((uint4*)vv.u32)[0] = vsrc[0];
      ((uint4*)vv.u32)[1] = vsrc[1];
#pragma unroll
      for (int e = 0; e < 16; e++) vT[seg * 16 + e][row] = vv.u16[e];
    }
    WAIT_ASYNC0();
    __syncthreads();

    // Scores for two 16-key tiles: S = Q(16x64) . K^T, k-dim split 32+32.
    float sc[2][8];
#pragma unroll
    for (int tt = 0; tt < 2; tt++) {
      Frag kf0, kf1;
#pragma unroll
      for (int i = 0; i < 8; i++) {
        kf0.u32[i] = *(const unsigned int*)&ks[tt * 16 + r][(i < 4) ? (half * 8 + 2 * i) : (16 + half * 8 + 2 * (i - 4))];
        kf1.u32[i] = *(const unsigned int*)&ks[tt * 16 + r][32 + ((i < 4) ? (half * 8 + 2 * i) : (16 + half * 8 + 2 * (i - 4)))];
      }
      v8f s = vzero();
      s = WMMA_BF16(qa[0].v, kf0.v, s);
      s = WMMA_BF16(qa[1].v, kf1.v, s);
#pragma unroll
      for (int j = 0; j < 8; j++) sc[tt][j] = s[j] * 0.125f;  // 1/sqrt(64)
    }

    // Online softmax: row lives in one 16-lane half at fixed accumulator slot.
    float mn[8], al[8];
#pragma unroll
    for (int j = 0; j < 8; j++) {
      float v = fmaxf(sc[0][j], sc[1][j]);
      v = fmaxf(v, __shfl_xor(v, 1, 32));
      v = fmaxf(v, __shfl_xor(v, 2, 32));
      v = fmaxf(v, __shfl_xor(v, 4, 32));
      v = fmaxf(v, __shfl_xor(v, 8, 32));
      mn[j] = fmaxf(mr[j], v);
      al[j] = __expf(mr[j] - mn[j]);
      mr[j] = mn[j];
    }
#pragma unroll
    for (int j = 0; j < 8; j++) {
      const float p0 = __expf(sc[0][j] - mn[j]);
      const float p1 = __expf(sc[1][j] - mn[j]);
      pl[w][half * 8 + j][r] = f2bf(p0);
      pl[w][half * 8 + j][16 + r] = f2bf(p1);
      float su = p0 + p1;
      su += __shfl_xor(su, 1, 32);
      su += __shfl_xor(su, 2, 32);
      su += __shfl_xor(su, 4, 32);
      su += __shfl_xor(su, 8, 32);
      Ls[j] = Ls[j] * al[j] + su;
#pragma unroll
      for (int nt = 0; nt < 4; nt++) ctx[nt][j] *= al[j];
    }

    // PV: ctx(16x64) += P(16x32) . V(32x64); P A-fragment via LDS reshape.
    Frag pa;
#pragma unroll
    for (int i = 0; i < 8; i++) {
      const int k0 = (i < 4) ? (half * 8 + 2 * i) : (16 + half * 8 + 2 * (i - 4));
      pa.u32[i] = *(const unsigned int*)&pl[w][r][k0];
    }
#pragma unroll
    for (int nt = 0; nt < 4; nt++) {
      Frag vf;
#pragma unroll
      for (int i = 0; i < 8; i++)
        vf.u32[i] = *(const unsigned int*)&vT[nt * 16 + r][(i < 4) ? (half * 8 + 2 * i) : (16 + half * 8 + 2 * (i - 4))];
      ctx[nt] = WMMA_BF16(pa.v, vf.v, ctx[nt]);
    }
    __syncthreads();
  }

  // Normalize and scatter rows through hilbert_map into (B,S,H) bf16 layout.
#pragma unroll
  for (int j = 0; j < 8; j++) {
    const float inv = 1.0f / Ls[j];
    const int s = qbase + half * 8 + j;
    const int sp = hm[s];
    unsigned short* dst = ctx_perm + ((size_t)b * 4096 + sp) * 1024 + h * 64;
#pragma unroll
    for (int nt = 0; nt < 4; nt++) dst[nt * 16 + r] = f2bf(ctx[nt][j] * inv);
  }
}

// ---------------------------------------------------------------------------
// Host-side launch
// ---------------------------------------------------------------------------
extern "C" void kernel_launch(void* const* d_in, const int* in_sizes, int n_in,
                              void* d_out, int out_size, void* d_ws, size_t ws_size,
                              hipStream_t stream) {
  (void)in_sizes; (void)n_in; (void)out_size; (void)ws_size;
  const float* x    = (const float*)d_in[0];   // [2,4096,1024]
  const float* wqkv = (const float*)d_in[1];   // [3072,1024]
  const float* wout = (const float*)d_in[2];   // [1024,1024]
  const int*   hm   = (const int*)d_in[3];     // [4096]
  float* out = (float*)d_out;                  // [2,4096,1024]

  char* ws = (char*)d_ws;
  auto carve = [&](size_t bytes) -> void* {
    void* p = (void*)ws;
    ws += (bytes + 255) & ~(size_t)255;
    return p;
  };
  const size_t NX = 2ull * 4096 * 1024;   // 8,388,608
  const size_t NWQ = 3072ull * 1024;      // 3,145,728
  const size_t NWO = 1024ull * 1024;      // 1,048,576
  unsigned short* xb  = (unsigned short*)carve(NX * 2);
  unsigned short* wqb = (unsigned short*)carve(NWQ * 2);
  unsigned short* wob = (unsigned short*)carve(NWO * 2);
  unsigned short* qb  = (unsigned short*)carve(NX * 2);
  unsigned short* kb  = (unsigned short*)carve(NX * 2);
  unsigned short* vb  = (unsigned short*)carve(NX * 2);
  unsigned short* ctxp = (unsigned short*)carve(NX * 2);
  int* hk = (int*)carve(1024 * 4);

  // 1) fp32 -> bf16 staging
  cvt_f32_bf16<<<(int)((NX + 255) / 256), 256, 0, stream>>>(x, xb, (int)NX);
  cvt_f32_bf16<<<(int)((NWQ + 255) / 256), 256, 0, stream>>>(wqkv, wqb, (int)NWQ);
  cvt_f32_bf16<<<(int)((NWO + 255) / 256), 256, 0, stream>>>(wout, wob, (int)NWO);

  // 2) gathered key index table
  build_hk<<<1, 1024, 0, stream>>>(hm, hk);

  // 3) QKV projection GEMM (8192 x 3072 x 1024), scatter into q/k/v
  gemm_qkv<<<dim3(64, 24), 256, 0, stream>>>(xb, wqb, qb, kb, vb);

  // 4) fused attention over 1024 selected keys, scatter via hilbert_map
  attn_fused<<<dim3(64, 32), 128, 0, stream>>>(qb, kb, vb, hk, hm, ctxp);

  // 5) output projection GEMM (8192 x 1024 x 1024) -> fp32 d_out
  gemm_out<<<dim3(64, 8), 256, 0, stream>>>(ctxp, wob, out);
}
